// GINE_83846351552683
// MI455X (gfx1250) — matrix operand order repaired and checked by reading, not compiled, not run on hardware
//
#include <hip/hip_runtime.h>

// ---------------------------------------------------------------------------
// GINE forward for MI455X (gfx1250, wave32, WMMA), problem constants:
#define N_NODES 50000
#define N_EDGES 1600000
#define N_GRAPH 512
#define DIM     64     // node feature dim (D_IN == H == 64)
#define EDIM    16     // edge attr dim
#define NCLS    2
#define TITER   4      // edge tiles (of 16 edges) per wave

typedef __attribute__((ext_vector_type(16))) __bf16 v16bf;
typedef __attribute__((ext_vector_type(8)))  float  v8f;

union BF16x16 {
  v16bf v;
  __bf16 h[16];
  uint4  q[2];
};

// Load a pre-swizzled bf16 B-tile: 32 B per lane, coalesced (2x b128).
__device__ __forceinline__ void loadB(const __bf16* __restrict__ swz,
                                      int tile_id, int lane, BF16x16& B) {
  const uint4* p = (const uint4*)(swz + ((unsigned)tile_id * 32 + lane) * 16);
  B.q[0] = p[0];
  B.q[1] = p[1];
}

// ---------------------------------------------------------------------------
// Weight pre-swizzle kernels: f32 row-major -> bf16 WMMA-B lane layout.
// B layout (32x16, 16-bit), matching the ISA A layout:
// element e within a lane -> K = 8*(lane>>4) + (e&7) + 16*(e>>3), N = 16*t + (lane&15).

// 64x64 weight -> 8 tiles (tile_id = cc*4 + t; cc = K-chunk of 32, t = N-block)
__global__ __launch_bounds__(256) void prep_w64(const float* __restrict__ W,
                                                __bf16* __restrict__ swz) {
  const int tid = blockIdx.x * 256 + threadIdx.x;      // 4096 threads
  const int tile_id = tid >> 9;
  const int l = (tid >> 4) & 31;
  const int e = tid & 15;
  const int cc = tile_id >> 2, t = tile_id & 3;
  const int K = 32 * cc + 8 * (l >> 4) + (e & 7) + 16 * (e >> 3);
  const int N = 16 * t + (l & 15);
  swz[tid] = (__bf16)W[K * DIM + N];
}

// 16x64 edge weight -> 4 tiles, K zero-padded 16 -> 32
__global__ __launch_bounds__(256) void prep_we(const float* __restrict__ We,
                                               __bf16* __restrict__ swz) {
  const int tid = blockIdx.x * 256 + threadIdx.x;      // 2048 threads
  const int t = tid >> 9;
  const int l = (tid >> 4) & 31;
  const int e = tid & 15;
  const int g = l >> 4;                                // e>=8 -> K in 16..31 -> pad
  const int N = 16 * t + (l & 15);
  swz[tid] = (e < 8) ? (__bf16)We[(8 * g + e) * DIM + N] : (__bf16)0.0f;
}

// ---------------------------------------------------------------------------
// Fused GINE message + scatter:  agg[dst] += relu(x[src] + edge_attr@We + be)
// One wave: loads 4 We B-tiles once, then processes TITER tiles of 16 edges.
// ISA 16-bit A layout: lane holds M=lane&15; element e -> K = e%8 + 8g + 16*(e/8).
// Gathers/atomics use unsigned 32-bit element offsets so the backend emits the
// GVS saddr+voffset form (1 VALU op per address instead of 64-bit chains).
__global__ __launch_bounds__(256) void edge_msg_scatter(
    const float* __restrict__ xin, const float* __restrict__ eattr,
    const int* __restrict__ src, const int* __restrict__ dst,
    const __bf16* __restrict__ swzWe, const float* __restrict__ be,
    float* __restrict__ agg)
{
  const int lane = threadIdx.x & 31;
  const int wave = threadIdx.x >> 5;
  const long base = ((long)blockIdx.x * 8 + wave) * (16 * TITER);
  const int row = lane & 15;
  const int g   = lane >> 4;

  BF16x16 B[4];
  float bias[4];
#pragma unroll
  for (int t = 0; t < 4; ++t) {
    loadB(swzWe, t, lane, B[t]);
    bias[t] = be[16 * t + row];
  }

#pragma unroll
  for (int it = 0; it < TITER; ++it) {
    const long e0 = base + it * 16;

    // A: this lane loads 8 contiguous attrs of edge (e0+row): K in [8g, 8g+7]
    const float* ea = eattr + (size_t)(e0 + row) * EDIM + g * 8;
    __builtin_prefetch(ea + 16 * EDIM, 0, 0);          // next tile (global_prefetch)
    const float4 a0 = ((const float4*)ea)[0];
    const float4 a1 = ((const float4*)ea)[1];
    BF16x16 A;
    A.h[0] = (__bf16)a0.x; A.h[1] = (__bf16)a0.y;
    A.h[2] = (__bf16)a0.z; A.h[3] = (__bf16)a0.w;
    A.h[4] = (__bf16)a1.x; A.h[5] = (__bf16)a1.y;
    A.h[6] = (__bf16)a1.z; A.h[7] = (__bf16)a1.w;
#pragma unroll
    for (int e = 8; e < 16; ++e) A.h[e] = (__bf16)0.0f;  // K >= 16 : zero pad

    // 32-bit element offsets for the 8 C-rows this half-wave owns (M = r + 8g);
    // per-t feature offset (16t floats) folds into the instruction IOFFSET.
    unsigned xoff[8], aoff[8];
#pragma unroll
    for (int r = 0; r < 8; ++r) {
      const int m = r + 8 * g;
      xoff[r] = (unsigned)src[e0 + m] * DIM + (unsigned)row;
      aoff[r] = (unsigned)dst[e0 + m] * DIM + (unsigned)row;
    }

    // Issue all 32 gathers up-front; they overlap with the WMMAs below.
    float xv[4][8];
#pragma unroll
    for (int t = 0; t < 4; ++t)
#pragma unroll
      for (int r = 0; r < 8; ++r) xv[t][r] = xin[xoff[r] + 16u * t];

#pragma unroll
    for (int t = 0; t < 4; ++t) {
      v8f c = {};
      c = __builtin_amdgcn_wmma_f32_16x16x32_bf16(false, A.v, false, B[t].v,
                                                  (short)0, c, false, false);
#pragma unroll
      for (int r = 0; r < 8; ++r) {
        const float m = fmaxf(c[r] + bias[t] + xv[t][r], 0.0f);
        atomicAdd(&agg[aoff[r] + 16u * t], m);         // L2-resident f32 atomics
      }
    }
  }
}

// ---------------------------------------------------------------------------
// Node MLP:  out = [relu]( relu((x+agg) @ W1 + b1) @ W2 + b2 )
// One wave = 16 nodes. 64x64 GEMM = 4 N-tiles x 2 K-chunks of WMMA.
// Hidden activations re-swizzled C-layout -> A-layout through LDS.
#define LSTRIDE (DIM + 4)   // 68 floats: keeps 16B alignment for b128 LDS loads

__global__ __launch_bounds__(256) void node_mlp(
    const float* __restrict__ xin, const float* __restrict__ agg,
    const __bf16* __restrict__ swzW1, const float* __restrict__ b1,
    const __bf16* __restrict__ swzW2, const float* __restrict__ b2,
    float* __restrict__ out, int relu_out)
{
  __shared__ float sh[8][16][LSTRIDE];
  const int lane = threadIdx.x & 31;
  const int wave = threadIdx.x >> 5;
  const int tile = blockIdx.x * 8 + wave;
  if (tile * 16 >= N_NODES) return;                    // wave-uniform exit
  const int row = lane & 15;
  const int g   = lane >> 4;
  const unsigned r0 = (unsigned)(tile * 16 + row) * DIM;

  // A chunks of h = xin + agg (K=0..31, 32..63), ISA 16-bit A layout:
  // elems 0-7 -> K = 32cc + 8g + e ; elems 8-15 -> K = 32cc + 16 + 8g + (e-8)
  BF16x16 Ac[2];
#pragma unroll
  for (int cc = 0; cc < 2; ++cc) {
    const float4* p1 = (const float4*)(xin + r0 + 32 * cc + 8 * g);
    const float4* p2 = (const float4*)(agg + r0 + 32 * cc + 8 * g);
    const float4 x0 = p1[0], x1 = p1[1], x2 = p1[4], x3 = p1[5];
    const float4 y0 = p2[0], y1 = p2[1], y2 = p2[4], y3 = p2[5];
    Ac[cc].h[0]  = (__bf16)(x0.x + y0.x); Ac[cc].h[1]  = (__bf16)(x0.y + y0.y);
    Ac[cc].h[2]  = (__bf16)(x0.z + y0.z); Ac[cc].h[3]  = (__bf16)(x0.w + y0.w);
    Ac[cc].h[4]  = (__bf16)(x1.x + y1.x); Ac[cc].h[5]  = (__bf16)(x1.y + y1.y);
    Ac[cc].h[6]  = (__bf16)(x1.z + y1.z); Ac[cc].h[7]  = (__bf16)(x1.w + y1.w);
    Ac[cc].h[8]  = (__bf16)(x2.x + y2.x); Ac[cc].h[9]  = (__bf16)(x2.y + y2.y);
    Ac[cc].h[10] = (__bf16)(x2.z + y2.z); Ac[cc].h[11] = (__bf16)(x2.w + y2.w);
    Ac[cc].h[12] = (__bf16)(x3.x + y3.x); Ac[cc].h[13] = (__bf16)(x3.y + y3.y);
    Ac[cc].h[14] = (__bf16)(x3.z + y3.z); Ac[cc].h[15] = (__bf16)(x3.w + y3.w);
  }

  // GEMM1 + bias + ReLU -> LDS (row-major per node)
#pragma unroll
  for (int t = 0; t < 4; ++t) {
    v8f acc = {};
#pragma unroll
    for (int cc = 0; cc < 2; ++cc) {
      BF16x16 B;
      loadB(swzW1, cc * 4 + t, lane, B);
      acc = __builtin_amdgcn_wmma_f32_16x16x32_bf16(false, Ac[cc].v, false, B.v,
                                                    (short)0, acc, false, false);
    }
    const int f = 16 * t + row;                        // C: N = lane&15
    const float bias = b1[f];
#pragma unroll
    for (int r = 0; r < 8; ++r)                        // C: M = r + 8g
      sh[wave][r + 8 * g][f] = fmaxf(acc[r] + bias, 0.0f);
  }

  // Reload hidden as A-layout (same-wave LDS; compiler orders DS ops)
  BF16x16 A2[2];
#pragma unroll
  for (int cc = 0; cc < 2; ++cc) {
    const float4* p = (const float4*)(&sh[wave][row][32 * cc + 8 * g]);
    const float4 x0 = p[0], x1 = p[1], x2 = p[4], x3 = p[5];
    A2[cc].h[0]  = (__bf16)x0.x; A2[cc].h[1]  = (__bf16)x0.y;
    A2[cc].h[2]  = (__bf16)x0.z; A2[cc].h[3]  = (__bf16)x0.w;
    A2[cc].h[4]  = (__bf16)x1.x; A2[cc].h[5]  = (__bf16)x1.y;
    A2[cc].h[6]  = (__bf16)x1.z; A2[cc].h[7]  = (__bf16)x1.w;
    A2[cc].h[8]  = (__bf16)x2.x; A2[cc].h[9]  = (__bf16)x2.y;
    A2[cc].h[10] = (__bf16)x2.z; A2[cc].h[11] = (__bf16)x2.w;
    A2[cc].h[12] = (__bf16)x3.x; A2[cc].h[13] = (__bf16)x3.y;
    A2[cc].h[14] = (__bf16)x3.z; A2[cc].h[15] = (__bf16)x3.w;
  }

  // GEMM2 + bias (+ optional outer ReLU), store f32
#pragma unroll
  for (int t = 0; t < 4; ++t) {
    v8f acc = {};
#pragma unroll
    for (int cc = 0; cc < 2; ++cc) {
      BF16x16 B;
      loadB(swzW2, cc * 4 + t, lane, B);
      acc = __builtin_amdgcn_wmma_f32_16x16x32_bf16(false, A2[cc].v, false, B.v,
                                                    (short)0, acc, false, false);
    }
    const int f = 16 * t + row;
    const float bias = b2[f];
    const unsigned obase = (unsigned)(tile * 16 + 8 * g) * DIM + (unsigned)f;
#pragma unroll
    for (int r = 0; r < 8; ++r) {                      // + r*DIM as IOFFSET
      float v = acc[r] + bias;
      if (relu_out) v = fmaxf(v, 0.0f);
      out[obase + (unsigned)r * DIM] = v;
    }
  }
}

// ---------------------------------------------------------------------------
__global__ __launch_bounds__(256) void pool_kernel(
    const float* __restrict__ h, const int* __restrict__ batch,
    float* __restrict__ sums, float* __restrict__ cnts)
{
  const unsigned idx = blockIdx.x * 256 + threadIdx.x; // exact: N*64 / 256 blocks
  const unsigned n = idx >> 6, f = idx & 63;
  const unsigned gph = (unsigned)batch[n];
  atomicAdd(&sums[gph * DIM + f], h[idx]);
  if (f == 0) atomicAdd(&cnts[gph], 1.0f);
}

__global__ __launch_bounds__(256) void readout1(
    const float* __restrict__ sums, const float* __restrict__ cnts,
    const float* __restrict__ Wl, const float* __restrict__ bl,
    float* __restrict__ z)
{
  const unsigned idx = blockIdx.x * 256 + threadIdx.x; // G*256 threads
  const unsigned gph = idx >> 8, j = idx & 255;
  const float inv = 1.0f / fmaxf(cnts[gph], 1.0f);
  float acc = bl[j];
#pragma unroll 8
  for (unsigned k = 0; k < DIM; ++k)
    acc += sums[gph * DIM + k] * inv * Wl[k * 256 + j];
  z[idx] = fmaxf(acc, 0.0f);
}

__global__ __launch_bounds__(256) void readout2(
    const float* __restrict__ z, const float* __restrict__ Wl2,
    const float* __restrict__ bl2, float* __restrict__ out)
{
  const unsigned idx = blockIdx.x * 256 + threadIdx.x;
  if (idx >= N_GRAPH * NCLS) return;
  const unsigned gph = idx >> 1, c = idx & 1;
  float acc = bl2[c];
#pragma unroll 8
  for (unsigned k = 0; k < 256; ++k)
    acc += z[gph * 256 + k] * Wl2[k * NCLS + c];
  out[idx] = acc;
}

// ---------------------------------------------------------------------------
extern "C" void kernel_launch(void* const* d_in, const int* in_sizes, int n_in,
                              void* d_out, int out_size, void* d_ws, size_t ws_size,
                              hipStream_t stream) {
  (void)in_sizes; (void)n_in; (void)out_size; (void)ws_size;
  const float* x     = (const float*)d_in[0];
  const int*   ei    = (const int*)d_in[1];
  const float* eattr = (const float*)d_in[2];
  const int*   batch = (const int*)d_in[3];
  const float* P[22];
  for (int i = 0; i < 22; ++i) P[i] = (const float*)d_in[4 + i];
  // P: 0 We1,1 be1,2 W11,3 b11,4 W12,5 b12, 6 We2,7 be2,8 W21,9 b21,10 W22,11 b22,
  //    12 We3,13 be3,14 W31,15 b31,16 W32,17 b32, 18 Wl,19 bl,20 Wl2,21 bl2
  const int* src = ei;
  const int* dst = ei + N_EDGES;

  float* ws   = (float*)d_ws;
  float* hA   = ws;                                   // [N,64]
  float* agg  = hA  + (size_t)N_NODES * DIM;          // [N,64]
  float* hB   = agg + (size_t)N_NODES * DIM;          // [N,64]
  float* sums = hB  + (size_t)N_NODES * DIM;          // [G,64]
  float* cnts = sums + (size_t)N_GRAPH * DIM;         // [G]
  float* z    = cnts + N_GRAPH;                       // [G,256]
  // bf16 swizzled-weight region (16B aligned: float offset is a multiple of 4)
  __bf16* swz  = (__bf16*)(z + (size_t)N_GRAPH * 256);
  __bf16* sWe1 = swz;              // 3 x 2048
  __bf16* sWe2 = sWe1 + 2048;
  __bf16* sWe3 = sWe2 + 2048;
  __bf16* sW11 = sWe3 + 2048;      // 6 x 4096
  __bf16* sW12 = sW11 + 4096;
  __bf16* sW21 = sW12 + 4096;
  __bf16* sW22 = sW21 + 4096;
  __bf16* sW31 = sW22 + 4096;
  __bf16* sW32 = sW31 + 4096;

  const dim3 blk(256);
  const int edgeBlocks = N_EDGES / (16 * 8 * TITER);  // 3125, exact
  const int mlpBlocks  = (N_NODES / 16 + 7) / 8;      // 391
  const int poolBlocks = (N_NODES * DIM) / 256;       // 12500, exact
  const size_t aggBytes = (size_t)N_NODES * DIM * sizeof(float);

  // ---- one-time weight pre-swizzle (bf16, WMMA-B lane layout) ----
  prep_we<<<8,  blk, 0, stream>>>(P[0],  sWe1);
  prep_we<<<8,  blk, 0, stream>>>(P[6],  sWe2);
  prep_we<<<8,  blk, 0, stream>>>(P[12], sWe3);
  prep_w64<<<16, blk, 0, stream>>>(P[2],  sW11);
  prep_w64<<<16, blk, 0, stream>>>(P[4],  sW12);
  prep_w64<<<16, blk, 0, stream>>>(P[8],  sW21);
  prep_w64<<<16, blk, 0, stream>>>(P[10], sW22);
  prep_w64<<<16, blk, 0, stream>>>(P[14], sW31);
  prep_w64<<<16, blk, 0, stream>>>(P[16], sW32);

  // ---- layer 1 (input = x) ----
  hipMemsetAsync(agg, 0, aggBytes, stream);
  edge_msg_scatter<<<edgeBlocks, blk, 0, stream>>>(x, eattr, src, dst, sWe1, P[1], agg);
  node_mlp<<<mlpBlocks, blk, 0, stream>>>(x, agg, sW11, P[3], sW12, P[5], hA, 1);
  // ---- layer 2 ----
  hipMemsetAsync(agg, 0, aggBytes, stream);
  edge_msg_scatter<<<edgeBlocks, blk, 0, stream>>>(hA, eattr, src, dst, sWe2, P[7], agg);
  node_mlp<<<mlpBlocks, blk, 0, stream>>>(hA, agg, sW21, P[9], sW22, P[11], hB, 1);
  // ---- layer 3 (no outer relu) ----
  hipMemsetAsync(agg, 0, aggBytes, stream);
  edge_msg_scatter<<<edgeBlocks, blk, 0, stream>>>(hB, eattr, src, dst, sWe3, P[13], agg);
  node_mlp<<<mlpBlocks, blk, 0, stream>>>(hB, agg, sW31, P[15], sW32, P[17], hA, 0);
  // ---- pool + head ----
  hipMemsetAsync(sums, 0, ((size_t)N_GRAPH * DIM + N_GRAPH) * sizeof(float), stream);
  pool_kernel<<<poolBlocks, blk, 0, stream>>>(hA, batch, sums, cnts);
  readout1<<<(N_GRAPH * 256) / 256, blk, 0, stream>>>(sums, cnts, P[18], P[19], z);
  readout2<<<(N_GRAPH * NCLS + 255) / 256, blk, 0, stream>>>(z, P[20], P[21], (float*)d_out);
}
